// SolarisMPBlock_14903536517453
// MI455X (gfx1250) — compile-verified, matrix operand order checked by reading, not run on hardware
//
#include <hip/hip_runtime.h>
#include <hip/hip_bf16.h>
#include <stdint.h>

// ---------------------------------------------------------------------------
// Problem constants (reference: B=1, NH=12, DIM=1536, HD=128, F=8,H=22,W=20,P=1)
// ---------------------------------------------------------------------------
#define T_TOK 3520
#define DIMC  1536
#define NHEAD 12
#define HDIM  128
#define BLKC  440      // S*P = H*W*P : block-causal block size
#define CF    22       // rope f-cols
#define CHW   21       // rope h-cols / w-cols

typedef __attribute__((ext_vector_type(16))) __bf16 v16bf;
typedef __attribute__((ext_vector_type(2)))  __bf16 v2bf;
typedef __attribute__((ext_vector_type(8)))  float  v8f;

// ---------------------------------------------------------------------------
// CDNA5 async global->LDS DMA (ASYNCcnt-tracked). INST_OFFSET is added to BOTH
// the per-lane global address and the per-lane LDS address (ISA 08 §4.4), so
// contiguous row chunks need one base + literal offsets.
// ---------------------------------------------------------------------------
__device__ __forceinline__ uint32_t lds32(const void* p) {
  return (uint32_t)(uintptr_t)p;   // low 32 bits of shared-aperture flat addr
}
#define ASYNC_LD_B128(ldsoff, gptr, OFF)                                     \
  asm volatile("global_load_async_to_lds_b128 %0, %1, off offset:" #OFF      \
               :: "v"(ldsoff), "v"(gptr) : "memory")
#define WAIT_ASYNC0() asm volatile("s_wait_asynccnt 0x0" ::: "memory")

// ---------------------------------------------------------------------------
// WMMA helpers (CDNA5 wave32, 16x16x32 bf16 -> f32)
// ---------------------------------------------------------------------------
__device__ __forceinline__ v8f wmma_bf16(v16bf a, v16bf b, v8f c) {
  return __builtin_amdgcn_wmma_f32_16x16x32_bf16(
      /*neg_a=*/false, a, /*neg_b=*/false, b,
      /*c_mod=*/(short)0, c, /*reuse_a=*/false, /*reuse_b=*/false);
}

// A/B fragment from row-major bf16 LDS tile. Lane l (0..15) reads logical row
// (lane&15); half = lane>>4 selects K groups per ISA 16-bit A layout:
// vgpr i<4: K = half*8 + 2i,2i+1 ; vgpr i>=4: K = 16 + half*8 + 2(i-4),+1
__device__ __forceinline__ v16bf frag_ld_row(const __bf16* __restrict__ p, int ld) {
  const int lane = threadIdx.x & 31;
  const __bf16* r = p + (lane & 15) * ld + ((lane >> 4) * 8);
  v16bf f;
#pragma unroll
  for (int i = 0; i < 8; ++i) {
    const int kk = (i < 4) ? (2 * i) : (16 + 2 * (i - 4));
    v2bf t = *(const v2bf*)(r + kk);           // 4B-aligned (even element offs)
    f[2 * i] = t[0];
    f[2 * i + 1] = t[1];
  }
  return f;
}

// B fragment where logical (k,n) lives at memory [k*ld + n] (e.g. V tile):
// lane indexes n contiguously, K walks rows.
__device__ __forceinline__ v16bf frag_ld_col(const __bf16* __restrict__ p, int ld) {
  const int lane = threadIdx.x & 31;
  const __bf16* c = p + (lane & 15) + ((lane >> 4) * 8) * ld;
  v16bf f;
#pragma unroll
  for (int i = 0; i < 8; ++i) {
    const int kk = (i < 4) ? (2 * i) : (16 + 2 * (i - 4));
    f[2 * i]     = c[kk * ld];
    f[2 * i + 1] = c[(kk + 1) * ld];
  }
  return f;
}

// ---------------------------------------------------------------------------
// fp32 -> bf16 convert
// ---------------------------------------------------------------------------
__global__ void cvt_f32_bf16(const float* __restrict__ x, __bf16* __restrict__ y, int n) {
  int i = blockIdx.x * blockDim.x + threadIdx.x;
  if (i < n) y[i] = (__bf16)x[i];
}

// W[k][n] fp32  ->  Wt[n][k] bf16  (tiled transpose)
__global__ void transpose_to_bf16(const float* __restrict__ W, __bf16* __restrict__ Wt,
                                  int K, int N) {
  __shared__ float t[32][33];
  const int kb = blockIdx.x * 32, nb = blockIdx.y * 32;
  const int tx = threadIdx.x, ty = threadIdx.y;   // 32 x 8
#pragma unroll
  for (int i = 0; i < 32; i += 8) {
    int kk = kb + ty + i, nn = nb + tx;
    t[ty + i][tx] = (kk < K && nn < N) ? W[(size_t)kk * N + nn] : 0.f;
  }
  __syncthreads();
#pragma unroll
  for (int i = 0; i < 32; i += 8) {
    int nn = nb + ty + i, kk = kb + tx;
    if (nn < N && kk < K) Wt[(size_t)nn * K + kk] = (__bf16)t[tx][ty + i];
  }
}

// ---------------------------------------------------------------------------
// WMMA GEMM:  C_f32[M x N] = A_bf16[M x K] * Bt_bf16[N x K]^T + bias
// 128x128 block tile, K-step 32, 8 waves of 32x64 (2x4 wmma tiles each).
// Double-buffered LDS filled by async DMA; DMA of tile k+1 overlaps compute
// of tile k. LDS stride 34 bf16 (68 B = 17 banks) -> conflict-free frag reads.
// ---------------------------------------------------------------------------
#define GBM 128
#define GBN 128
#define GBK 32
#define GLD 34

__global__ __launch_bounds__(256)
void gemm_bf16_f32(const __bf16* __restrict__ A, const __bf16* __restrict__ Bt,
                   const float* __restrict__ bias, float* __restrict__ C,
                   int M, int N, int K) {
  __shared__ __bf16 As[2][GBM * GLD];
  __shared__ __bf16 Bs[2][GBN * GLD];

  const int tid = threadIdx.x;
  const int wid = tid >> 5;
  const int lane = tid & 31;
  const int m0 = blockIdx.y * GBM;
  const int n0 = blockIdx.x * GBN;

  const int wm = (wid >> 1) * 32;   // 0,32,64,96
  const int wn = (wid & 1) * 64;    // 0,64

  v8f acc[2][4];
#pragma unroll
  for (int i = 0; i < 2; ++i)
#pragma unroll
    for (int j = 0; j < 4; ++j)
#pragma unroll
      for (int e = 0; e < 8; ++e) acc[i][j][e] = 0.f;

  const int row = tid >> 1;             // 0..127
  const int colb = (tid & 1) * 16;      // 0 / 16
  const int gm = m0 + row;
  const int gn = n0 + row;
  const bool am = gm < M;
  const bool bm = gn < N;
  const __bf16* ga = A + (size_t)(am ? gm : 0) * K + colb;
  const __bf16* gb = Bt + (size_t)(bm ? gn : 0) * K + colb;
  const uint32_t laA[2] = { lds32(&As[0][row * GLD + colb]),
                            lds32(&As[1][row * GLD + colb]) };
  const uint32_t laB[2] = { lds32(&Bs[0][row * GLD + colb]),
                            lds32(&Bs[1][row * GLD + colb]) };

  // zero-fill OOB rows once (k-invariant; async DMA never touches them)
  if (m0 + GBM > M || n0 + GBN > N) {
    for (int idx = tid; idx < GBM * GBK; idx += 256) {
      const int r = idx >> 5, c = idx & 31;
      if (m0 + r >= M) { As[0][r * GLD + c] = (__bf16)0.f; As[1][r * GLD + c] = (__bf16)0.f; }
      if (n0 + r >= N) { Bs[0][r * GLD + c] = (__bf16)0.f; Bs[1][r * GLD + c] = (__bf16)0.f; }
    }
  }

  const int nk = K / GBK;
  // prologue: DMA tile 0 into buffer 0
  if (am) { ASYNC_LD_B128(laA[0], ga, 0); ASYNC_LD_B128(laA[0], ga, 16); }
  if (bm) { ASYNC_LD_B128(laB[0], gb, 0); ASYNC_LD_B128(laB[0], gb, 16); }
  WAIT_ASYNC0();
  __syncthreads();

  int cur = 0;
  for (int kt = 0; kt < nk; ++kt) {
    // issue DMA for next tile into the other buffer (overlaps wmma below)
    if (kt + 1 < nk) {
      const __bf16* gan = ga + (size_t)(kt + 1) * GBK;
      const __bf16* gbn = gb + (size_t)(kt + 1) * GBK;
      const int nb = 1 - cur;
      if (am) { ASYNC_LD_B128(laA[nb], gan, 0); ASYNC_LD_B128(laA[nb], gan, 16); }
      if (bm) { ASYNC_LD_B128(laB[nb], gbn, 0); ASYNC_LD_B128(laB[nb], gbn, 16); }
    }

    const __bf16* as = As[cur];
    const __bf16* bs = Bs[cur];
    v16bf af[2], bfr[4];
#pragma unroll
    for (int i = 0; i < 2; ++i) af[i] = frag_ld_row(as + (wm + 16 * i) * GLD, GLD);
#pragma unroll
    for (int j = 0; j < 4; ++j) bfr[j] = frag_ld_row(bs + (wn + 16 * j) * GLD, GLD);
#pragma unroll
    for (int i = 0; i < 2; ++i)
#pragma unroll
      for (int j = 0; j < 4; ++j) acc[i][j] = wmma_bf16(af[i], bfr[j], acc[i][j]);

    WAIT_ASYNC0();       // next tile landed
    __syncthreads();     // everyone done reading cur; next buffer visible
    cur ^= 1;
  }

  // ---- epilogue: C-layout store + bias ----
  const int cn = lane & 15;
  const int mb = (lane >> 4) * 8;
#pragma unroll
  for (int i = 0; i < 2; ++i) {
#pragma unroll
    for (int j = 0; j < 4; ++j) {
      const int gnn = n0 + wn + 16 * j + cn;
      const float bv = (gnn < N) ? bias[gnn] : 0.f;
#pragma unroll
      for (int r = 0; r < 8; ++r) {
        const int gmm = m0 + wm + 16 * i + mb + r;
        if (gmm < M && gnn < N) C[(size_t)gmm * N + gnn] = acc[i][j][r] + bv;
      }
    }
  }
}

// ---------------------------------------------------------------------------
// Q/K full-dim RMSNorm + 3D RoPE + bf16 pack; V bf16 pack. One block / token.
// ---------------------------------------------------------------------------
__global__ __launch_bounds__(256)
void qk_rope_v_pack(const float* __restrict__ qr, const float* __restrict__ kr,
                    const float* __restrict__ vr,
                    const float* __restrict__ gq, const float* __restrict__ gk,
                    const float* __restrict__ freqs,      // [512][64]
                    __bf16* __restrict__ qo, __bf16* __restrict__ ko,
                    __bf16* __restrict__ vo) {
  const int t = blockIdx.x;
  const int tid = threadIdx.x;
  __shared__ float red[2][8];
  __shared__ float cs[64], sn[64];
  __shared__ float rq_s, rk_s;

  if (tid < 64) {                                 // rope angle table for token t
    const int f = t / BLKC, s = t % BLKC;
    const int hh = s / 20, ww = s % 20;
    const int pos = (tid < CF) ? f : ((tid < CF + CHW) ? hh : ww);
    const float a = freqs[pos * 64 + tid];
    cs[tid] = __cosf(a);
    sn[tid] = __sinf(a);
  }

  const float* qrow = qr + (size_t)t * DIMC;
  const float* krow = kr + (size_t)t * DIMC;
  float sq = 0.f, sk = 0.f;
  for (int i = tid; i < DIMC; i += 256) {
    const float a = qrow[i]; sq += a * a;
    const float b = krow[i]; sk += b * b;
  }
#pragma unroll
  for (int m = 16; m >= 1; m >>= 1) {
    sq += __shfl_xor(sq, m, 32);
    sk += __shfl_xor(sk, m, 32);
  }
  if ((tid & 31) == 0) { red[0][tid >> 5] = sq; red[1][tid >> 5] = sk; }
  __syncthreads();
  if (tid == 0) {
    float a = 0.f, b = 0.f;
#pragma unroll
    for (int i = 0; i < 8; ++i) { a += red[0][i]; b += red[1][i]; }
    rq_s = rsqrtf(a / (float)DIMC + 1e-6f);
    rk_s = rsqrtf(b / (float)DIMC + 1e-6f);
  }
  __syncthreads();
  const float rq = rq_s, rk = rk_s;

  // 12 heads * 64 pairs = 768 pairs
  for (int pi = tid; pi < NHEAD * 64; pi += 256) {
    const int h = pi >> 6, j = pi & 63;
    const int i0 = h * HDIM + 2 * j, i1 = i0 + 1;
    const float c = cs[j], s = sn[j];
    const float q0 = qrow[i0] * rq * gq[i0], q1 = qrow[i1] * rq * gq[i1];
    qo[(size_t)t * DIMC + i0] = (__bf16)(q0 * c - q1 * s);
    qo[(size_t)t * DIMC + i1] = (__bf16)(q0 * s + q1 * c);
    const float k0 = krow[i0] * rk * gk[i0], k1 = krow[i1] * rk * gk[i1];
    ko[(size_t)t * DIMC + i0] = (__bf16)(k0 * c - k1 * s);
    ko[(size_t)t * DIMC + i1] = (__bf16)(k0 * s + k1 * c);
  }
  for (int i = tid; i < DIMC; i += 256)
    vo[(size_t)t * DIMC + i] = (__bf16)vr[(size_t)t * DIMC + i];
}

// ---------------------------------------------------------------------------
// Flash-style block-causal attention. Grid: (T/64, NH). 128 threads = 4 waves,
// each wave owns 16 query rows; 32-key steps; QK^T and P*V via WMMA.
// K/V tiles double-buffered via async DMA (DMA of step+1 overlaps compute).
// All K/V loads are in-bounds by construction (kend <= T).
// ---------------------------------------------------------------------------
#define ALQ 136   // 64x128 Q tile, padded (272B = 68 banks -> conflict-free)
#define ALK 136   // 32x128 K/V tiles
#define ALP 34    // per-wave 16x32 P tile

__global__ __launch_bounds__(128)
void attn_kernel(const __bf16* __restrict__ q, const __bf16* __restrict__ k,
                 const __bf16* __restrict__ v, __bf16* __restrict__ o) {
  __shared__ __bf16 Qs[64 * ALQ];
  __shared__ __bf16 Ks[2][32 * ALK];
  __shared__ __bf16 Vs[2][32 * ALK];
  __shared__ __bf16 Ps[4 * 16 * ALP];

  const int h  = blockIdx.y;
  const int q0 = blockIdx.x * 64;
  const int tid = threadIdx.x;
  const int wid = tid >> 5, lane = tid & 31;

  // ---- stage Q tile (64 x 128) via async DMA: 8 x b128 per thread ----
  {
    const int r = tid >> 1, cb = (tid & 1) * 64;
    const __bf16* g = q + (size_t)(q0 + r) * DIMC + h * HDIM + cb;
    const uint32_t l = lds32(Qs + r * ALQ + cb);
    ASYNC_LD_B128(l, g, 0);   ASYNC_LD_B128(l, g, 16);
    ASYNC_LD_B128(l, g, 32);  ASYNC_LD_B128(l, g, 48);
    ASYNC_LD_B128(l, g, 64);  ASYNC_LD_B128(l, g, 80);
    ASYNC_LD_B128(l, g, 96);  ASYNC_LD_B128(l, g, 112);
  }

  // per-thread K/V staging geometry: 4 x b128 per tile
  const int rkv = tid >> 2, cbkv = (tid & 3) * 32;
  const __bf16* gk0 = k + (size_t)rkv * DIMC + h * HDIM + cbkv;
  const __bf16* gv0 = v + (size_t)rkv * DIMC + h * HDIM + cbkv;
  const uint32_t lK[2] = { lds32(&Ks[0][rkv * ALK + cbkv]), lds32(&Ks[1][rkv * ALK + cbkv]) };
  const uint32_t lV[2] = { lds32(&Vs[0][rkv * ALK + cbkv]), lds32(&Vs[1][rkv * ALK + cbkv]) };

  // prologue: DMA K/V step 0 into buffer 0 (Q still in flight too)
  ASYNC_LD_B128(lK[0], gk0, 0);  ASYNC_LD_B128(lK[0], gk0, 16);
  ASYNC_LD_B128(lK[0], gk0, 32); ASYNC_LD_B128(lK[0], gk0, 48);
  ASYNC_LD_B128(lV[0], gv0, 0);  ASYNC_LD_B128(lV[0], gv0, 16);
  ASYNC_LD_B128(lV[0], gv0, 32); ASYNC_LD_B128(lV[0], gv0, 48);
  WAIT_ASYNC0();
  __syncthreads();

  v16bf qf[4];
#pragma unroll
  for (int d = 0; d < 4; ++d)
    qf[d] = frag_ld_row(Qs + wid * 16 * ALQ + d * 32, ALQ);

  v8f oacc[8];
#pragma unroll
  for (int i = 0; i < 8; ++i)
#pragma unroll
    for (int e = 0; e < 8; ++e) oacc[i][e] = 0.f;

  float mrow[8], lrow[8];
#pragma unroll
  for (int r = 0; r < 8; ++r) { mrow[r] = -1e30f; lrow[r] = 0.f; }

  const int mb = (lane >> 4) * 8;
  const int cn = lane & 15;
  int qfr[8];
#pragma unroll
  for (int r = 0; r < 8; ++r) qfr[r] = (q0 + wid * 16 + mb + r) / BLKC;

  const int fmax = (q0 + 63) / BLKC;
  const int kend = (fmax + 1) * BLKC;
  const int nsteps = (kend + 31) / 32;
  const float scale = 0.088388347648318447f;   // 1/sqrt(128)

  int cur = 0;
  for (int step = 0; step < nsteps; ++step) {
    const int kb = step * 32;
    // DMA next K/V step into the other buffer; overlaps the wmma work below
    if (step + 1 < nsteps) {
      const int nb = 1 - cur;
      const __bf16* gkn = gk0 + (size_t)(step + 1) * 32 * DIMC;
      const __bf16* gvn = gv0 + (size_t)(step + 1) * 32 * DIMC;
      ASYNC_LD_B128(lK[nb], gkn, 0);  ASYNC_LD_B128(lK[nb], gkn, 16);
      ASYNC_LD_B128(lK[nb], gkn, 32); ASYNC_LD_B128(lK[nb], gkn, 48);
      ASYNC_LD_B128(lV[nb], gvn, 0);  ASYNC_LD_B128(lV[nb], gvn, 16);
      ASYNC_LD_B128(lV[nb], gvn, 32); ASYNC_LD_B128(lV[nb], gvn, 48);
    }
    const __bf16* ks = Ks[cur];
    const __bf16* vs = Vs[cur];

    // ---- S = Q K^T : two 16(q) x 16(key) tiles, K-dim = HD in 4 chunks ----
    v8f sfr[2];
#pragma unroll
    for (int kt = 0; kt < 2; ++kt) {
      v8f a;
#pragma unroll
      for (int e = 0; e < 8; ++e) a[e] = 0.f;
#pragma unroll
      for (int d = 0; d < 4; ++d) {
        v16bf bfr = frag_ld_row(ks + (kt * 16) * ALK + d * 32, ALK);
        a = wmma_bf16(qf[d], bfr, a);
      }
      sfr[kt] = a;
    }

    // ---- mask + scale + online softmax (C-layout aware) ----
    float pm[2][8], rmax[8];
#pragma unroll
    for (int r = 0; r < 8; ++r) rmax[r] = -1e30f;
#pragma unroll
    for (int kt = 0; kt < 2; ++kt) {
      const int key = kb + kt * 16 + cn;
      const int kfr = key / BLKC;
#pragma unroll
      for (int r = 0; r < 8; ++r) {
        const float sv = (kfr <= qfr[r]) ? sfr[kt][r] * scale : -1e30f;
        pm[kt][r] = sv;
        rmax[r] = fmaxf(rmax[r], sv);
      }
    }
#pragma unroll
    for (int m = 1; m < 16; m <<= 1)
#pragma unroll
      for (int r = 0; r < 8; ++r) rmax[r] = fmaxf(rmax[r], __shfl_xor(rmax[r], m, 32));

    float alpha[8], rsum[8];
#pragma unroll
    for (int r = 0; r < 8; ++r) {
      const float mn = fmaxf(mrow[r], rmax[r]);
      alpha[r] = __expf(mrow[r] - mn);
      mrow[r] = mn;
      rsum[r] = 0.f;
    }

    __bf16* pw = Ps + wid * 16 * ALP;
#pragma unroll
    for (int kt = 0; kt < 2; ++kt)
#pragma unroll
      for (int r = 0; r < 8; ++r) {
        const float p = __expf(pm[kt][r] - mrow[r]);
        rsum[r] += p;
        pw[(mb + r) * ALP + kt * 16 + cn] = (__bf16)p;
      }
#pragma unroll
    for (int m = 1; m < 16; m <<= 1)
#pragma unroll
      for (int r = 0; r < 8; ++r) rsum[r] += __shfl_xor(rsum[r], m, 32);
#pragma unroll
    for (int r = 0; r < 8; ++r) lrow[r] = lrow[r] * alpha[r] + rsum[r];

#pragma unroll
    for (int nt = 0; nt < 8; ++nt)
#pragma unroll
      for (int r = 0; r < 8; ++r) oacc[nt][r] *= alpha[r];

    // ---- O += P * V ----
    v16bf pf = frag_ld_row(pw, ALP);        // per-wave region: no barrier needed
#pragma unroll
    for (int nt = 0; nt < 8; ++nt) {
      v16bf vf = frag_ld_col(vs + nt * 16, ALK);
      oacc[nt] = wmma_bf16(pf, vf, oacc[nt]);
    }

    WAIT_ASYNC0();       // next K/V landed
    __syncthreads();     // everyone done reading cur
    cur ^= 1;
  }

  // ---- finalize: O /= l, store bf16 head-concat layout [T][DIM] ----
#pragma unroll
  for (int nt = 0; nt < 8; ++nt)
#pragma unroll
    for (int r = 0; r < 8; ++r) {
      const float val = oacc[nt][r] / lrow[r];
      const int gm = q0 + wid * 16 + mb + r;
      o[(size_t)gm * DIMC + h * HDIM + nt * 16 + cn] = (__bf16)val;
    }
}

// ---------------------------------------------------------------------------
// Host launcher
// ---------------------------------------------------------------------------
static inline size_t alignup(size_t x) { return (x + 255) & ~(size_t)255; }

extern "C" void kernel_launch(void* const* d_in, const int* in_sizes, int n_in,
                              void* d_out, int out_size, void* d_ws, size_t ws_size,
                              hipStream_t stream) {
  const float* x     = (const float*)d_in[0];
  const float* freqs = (const float*)d_in[1];
  const float* Wq = (const float*)d_in[2];
  const float* bq = (const float*)d_in[3];
  const float* Wk = (const float*)d_in[4];
  const float* bk = (const float*)d_in[5];
  const float* Wv = (const float*)d_in[6];
  const float* bv = (const float*)d_in[7];
  const float* Wo = (const float*)d_in[8];
  const float* bo = (const float*)d_in[9];
  const float* gq = (const float*)d_in[10];
  const float* gk = (const float*)d_in[11];
  float* out = (float*)d_out;

  const size_t nTD = (size_t)T_TOK * DIMC;
  const size_t nWW = (size_t)DIMC * DIMC;

  char* p = (char*)d_ws;
  __bf16* xb  = (__bf16*)p; p += alignup(nTD * 2);
  __bf16* wqt = (__bf16*)p; p += alignup(nWW * 2);
  __bf16* wkt = (__bf16*)p; p += alignup(nWW * 2);
  __bf16* wvt = (__bf16*)p; p += alignup(nWW * 2);
  __bf16* wot = (__bf16*)p; p += alignup(nWW * 2);
  float*  qraw = (float*)p; p += alignup(nTD * 4);
  float*  kraw = (float*)p; p += alignup(nTD * 4);
  float*  vraw = (float*)p; p += alignup(nTD * 4);
  __bf16* qb = (__bf16*)p; p += alignup(nTD * 2);
  __bf16* kb2 = (__bf16*)p; p += alignup(nTD * 2);
  __bf16* vb2 = (__bf16*)p; p += alignup(nTD * 2);
  __bf16* ab = (__bf16*)p; p += alignup(nTD * 2);

  // 1) x -> bf16
  cvt_f32_bf16<<<(int)((nTD + 255) / 256), 256, 0, stream>>>(x, xb, (int)nTD);

  // 2) weights -> bf16 transposed (Wt[n][k] = W[k][n])
  {
    dim3 g(DIMC / 32, DIMC / 32), b(32, 8);
    transpose_to_bf16<<<g, b, 0, stream>>>(Wq, wqt, DIMC, DIMC);
    transpose_to_bf16<<<g, b, 0, stream>>>(Wk, wkt, DIMC, DIMC);
    transpose_to_bf16<<<g, b, 0, stream>>>(Wv, wvt, DIMC, DIMC);
    transpose_to_bf16<<<g, b, 0, stream>>>(Wo, wot, DIMC, DIMC);
  }

  // 3) Q/K/V projections (WMMA GEMM, f32 out)
  {
    dim3 g(DIMC / GBN, (T_TOK + GBM - 1) / GBM);   // (12, 28)
    gemm_bf16_f32<<<g, 256, 0, stream>>>(xb, wqt, bq, qraw, T_TOK, DIMC, DIMC);
    gemm_bf16_f32<<<g, 256, 0, stream>>>(xb, wkt, bk, kraw, T_TOK, DIMC, DIMC);
    gemm_bf16_f32<<<g, 256, 0, stream>>>(xb, wvt, bv, vraw, T_TOK, DIMC, DIMC);
  }

  // 4) RMSNorm + RoPE + bf16 pack
  qk_rope_v_pack<<<T_TOK, 256, 0, stream>>>(qraw, kraw, vraw, gq, gk, freqs,
                                            qb, kb2, vb2);

  // 5) block-causal flash attention (WMMA)
  {
    dim3 g(T_TOK / 64, NHEAD);   // (55, 12)
    attn_kernel<<<g, 128, 0, stream>>>(qb, kb2, vb2, ab);
  }

  // 6) output projection -> f32 d_out
  {
    dim3 g(DIMC / GBN, (T_TOK + GBM - 1) / GBM);
    gemm_bf16_f32<<<g, 256, 0, stream>>>(ab, wot, bo, out, T_TOK, DIMC, DIMC);
  }

  (void)in_sizes; (void)n_in; (void)out_size; (void)ws_size;
}
